// EncoderLayer_62500364091566
// MI455X (gfx1250) — compile-verified, matrix operand order checked by reading, not run on hardware
//
#include <hip/hip_runtime.h>
#include <math.h>

typedef __attribute__((ext_vector_type(16))) __bf16      v16bf;
typedef __attribute__((ext_vector_type(8)))  float        v8f;
typedef __attribute__((ext_vector_type(4)))  float        v4f;
typedef __attribute__((ext_vector_type(4)))  unsigned int v4u;
typedef __attribute__((ext_vector_type(8)))  unsigned int v8u;

#define BM 128
#define BN 128
#define BKK 32
#define KPAD 40   // bf16 elems per LDS row: pads banks, keeps every 8-elem chunk 16B aligned

// ---- helpers -------------------------------------------------------------
__device__ __forceinline__ unsigned short f2bf(float x) {
  unsigned int u = __float_as_uint(x);
  unsigned int r = (u + 0x7FFFu + ((u >> 16) & 1u)) >> 16;   // RNE
  return (unsigned short)r;
}
__device__ __forceinline__ unsigned int pack2bf(float a, float b) {
#if __has_builtin(__builtin_amdgcn_cvt_pk_bf16_f32)
  typedef __attribute__((ext_vector_type(2))) __bf16 v2bf;
  v2bf r = __builtin_amdgcn_cvt_pk_bf16_f32(a, b);
  return __builtin_bit_cast(unsigned int, r);
#else
  return (unsigned int)f2bf(a) | ((unsigned int)f2bf(b) << 16);
#endif
}
__device__ __forceinline__ float silu_f(float x) { return x / (1.f + __expf(-x)); }
__device__ __forceinline__ float softplus_f(float x) {
  return (x > 20.f) ? x : log1pf(__expf(x));
}
__device__ __forceinline__ float gelu_f(float x) {
  return 0.5f * x * (1.f + erff(x * 0.70710678118654752f));
}
// async DMA: global (16B) -> LDS, tracked by ASYNCcnt
__device__ __forceinline__ void async_ld_b128(unsigned int lds_off, const void* gptr) {
  asm volatile("global_load_async_to_lds_b128 %0, %1, off"
               :: "v"(lds_off), "v"((unsigned long long)(size_t)gptr)
               : "memory");
}

// ---- bf16-WMMA GEMM: out = A[M,K](bf16) * W[N,K](bf16)^T (+bias, epilogue) -
// C (f32, nullable), Cbf (bf16, nullable). epilogue: 0=none,1=softplus,2=gelu.
// Requires: M % 128 == 0, K % 32 == 0 (true for every call below).
// Double-buffered LDS; each wave issues exactly 4 async loads per K-tile, so
// "s_wait_asynccnt 4" overlaps next-tile DMA with current-tile WMMAs.
__global__ __launch_bounds__(256)
void gemm_bf16_wmma(const unsigned short* __restrict__ A, int lda,
                    const unsigned short* __restrict__ W, int ldw,
                    const float* __restrict__ bias,
                    float* __restrict__ C, unsigned short* __restrict__ Cbf,
                    int ldc, int M, int N, int K,
                    int epilogue, int accumulate) {
  __shared__ unsigned short sA[2][BM * KPAD];
  __shared__ unsigned short sB[2][BN * KPAD];

  const int tid   = threadIdx.x;
  const int lane  = tid & 31;
  const int wave  = tid >> 5;       // 0..7
  const int waveM = wave >> 1;      // 0..3 -> 32 rows each
  const int waveN = wave & 1;       // 0..1 -> 64 cols each
  const int blockM = blockIdx.y * BM;
  const int blockN = blockIdx.x * BN;

  v8f acc[2][4] = {};

  const int cBase = (lane >> 4) * 8;    // bf16 col base inside 32-wide K tile
  const int lm    = lane & 15;

  // per-thread staging coordinates (2 chunks of A, 2 chunks of W)
  const int rS0  = tid >> 2;                 // rows 0..63
  const int rS1  = (tid + 256) >> 2;         // rows 64..127
  const int chS  = (tid & 3) * 8;            // bf16 col base of 16B chunk
  int wr0 = blockN + rS0; if (wr0 > N - 1) wr0 = N - 1;   // clamp: cols>=N masked at store
  int wr1 = blockN + rS1; if (wr1 > N - 1) wr1 = N - 1;

  const int nk = K / BKK;

  // prologue: stage tile 0 into buffer 0
  {
    async_ld_b128((unsigned int)(size_t)&sA[0][rS0 * KPAD + chS],
                  A + (size_t)(blockM + rS0) * lda + chS);
    async_ld_b128((unsigned int)(size_t)&sA[0][rS1 * KPAD + chS],
                  A + (size_t)(blockM + rS1) * lda + chS);
    async_ld_b128((unsigned int)(size_t)&sB[0][rS0 * KPAD + chS],
                  W + (size_t)wr0 * ldw + chS);
    async_ld_b128((unsigned int)(size_t)&sB[0][rS1 * KPAD + chS],
                  W + (size_t)wr1 * ldw + chS);
  }

  for (int ki = 0; ki < nk; ++ki) {
    const int cur = ki & 1;
    if (ki + 1 < nk) {
      const int nxt = cur ^ 1;
      const int k1  = (ki + 1) * BKK;
      async_ld_b128((unsigned int)(size_t)&sA[nxt][rS0 * KPAD + chS],
                    A + (size_t)(blockM + rS0) * lda + k1 + chS);
      async_ld_b128((unsigned int)(size_t)&sA[nxt][rS1 * KPAD + chS],
                    A + (size_t)(blockM + rS1) * lda + k1 + chS);
      async_ld_b128((unsigned int)(size_t)&sB[nxt][rS0 * KPAD + chS],
                    W + (size_t)wr0 * ldw + k1 + chS);
      async_ld_b128((unsigned int)(size_t)&sB[nxt][rS1 * KPAD + chS],
                    W + (size_t)wr1 * ldw + k1 + chS);
      if (ki + 2 < nk)   // pre-warm L2 two tiles ahead (global_prefetch_b8)
        __builtin_prefetch(A + (size_t)(blockM + (tid >> 1)) * lda + (ki + 2) * BKK, 0, 1);
      asm volatile("s_wait_asynccnt 4" ::: "memory");   // tile ki landed; ki+1 in flight
    } else {
      asm volatile("s_wait_asynccnt 0" ::: "memory");
    }
    __syncthreads();

    // ---- fragments per ISA 16-bit A/B layout, 8 WMMAs ----
    v16bf afrag[2], bfrag[4];
#pragma unroll
    for (int mi = 0; mi < 2; ++mi) {
      int row = waveM * 32 + mi * 16 + lm;
      v4u a0 = *(const v4u*)(&sA[cur][row * KPAD + cBase]);
      v4u a1 = *(const v4u*)(&sA[cur][row * KPAD + cBase + 16]);
      v8u t = {a0.x, a0.y, a0.z, a0.w, a1.x, a1.y, a1.z, a1.w};
      afrag[mi] = __builtin_bit_cast(v16bf, t);
    }
#pragma unroll
    for (int ni = 0; ni < 4; ++ni) {
      int col = waveN * 64 + ni * 16 + lm;
      v4u b0 = *(const v4u*)(&sB[cur][col * KPAD + cBase]);
      v4u b1 = *(const v4u*)(&sB[cur][col * KPAD + cBase + 16]);
      v8u t = {b0.x, b0.y, b0.z, b0.w, b1.x, b1.y, b1.z, b1.w};
      bfrag[ni] = __builtin_bit_cast(v16bf, t);
    }
#pragma unroll
    for (int mi = 0; mi < 2; ++mi)
#pragma unroll
      for (int ni = 0; ni < 4; ++ni)
        acc[mi][ni] = __builtin_amdgcn_wmma_f32_16x16x32_bf16(
            false, afrag[mi], false, bfrag[ni], (short)0, acc[mi][ni], false, false);
    __syncthreads();   // buf cur free for tile ki+2's DMA next iteration
  }

  // ---- epilogue: C VGPR j -> (M = j or j+8 by lane half, N = lane%16) ----
#pragma unroll
  for (int mi = 0; mi < 2; ++mi) {
#pragma unroll
    for (int ni = 0; ni < 4; ++ni) {
      int col = blockN + waveN * 64 + ni * 16 + lm;
      if (col >= N) continue;
      int rowBase = blockM + waveM * 32 + mi * 16 + ((lane < 16) ? 0 : 8);
#pragma unroll
      for (int j = 0; j < 8; ++j) {
        int row = rowBase + j;
        float v = acc[mi][ni][j];
        if (bias) v += bias[col];
        if (accumulate && C) v += C[(size_t)row * ldc + col];
        if (epilogue == 1) v = softplus_f(v);
        else if (epilogue == 2) v = gelu_f(v);
        if (C)   C  [(size_t)row * ldc + col] = v;
        if (Cbf) Cbf[(size_t)row * ldc + col] = f2bf(v);
      }
    }
  }
}

// ---- elementwise f32 -> bf16 (8 elems/thread; n % 8 == 0) ------------------
__global__ __launch_bounds__(256)
void cvt_f32_bf16(const float* __restrict__ in, unsigned short* __restrict__ out, int n) {
  int i = (blockIdx.x * 256 + threadIdx.x) * 8;
  if (i >= n) return;
  v4f a = *(const v4f*)(in + i);
  v4f b = *(const v4f*)(in + i + 4);
  v4u r;
  r.x = pack2bf(a.x, a.y); r.y = pack2bf(a.z, a.w);
  r.z = pack2bf(b.x, b.y); r.w = pack2bf(b.z, b.w);
  *(v4u*)(out + i) = r;
}

// ---- depthwise causal conv (D_CONV=4) + SiLU; dir=1 runs in flipped space --
__global__ __launch_bounds__(256)
void conv_silu_kernel(const float* __restrict__ xz, const float* __restrict__ wconv,
                      const float* __restrict__ bconv, float* __restrict__ xc,
                      unsigned short* __restrict__ xcbf, int dir) {
  int idx = blockIdx.x * 256 + threadIdx.x;     // over 8*1024*1024
  if (idx >= 8 * 1024 * 1024) return;
  int c = idx & 1023;
  int t = (idx >> 10) & 1023;
  int b = idx >> 20;
  float s = bconv[c];
#pragma unroll
  for (int j = 0; j < 4; ++j) {
    int lsrc; bool ok;
    if (dir == 0) { lsrc = t - 3 + j;        ok = (lsrc >= 0); }
    else          { lsrc = 1023 - t + 3 - j; ok = (lsrc <= 1023); }
    if (ok) s += wconv[c * 4 + j] * xz[((size_t)(b * 1024 + lsrc)) * 2048 + c];
  }
  float r = silu_f(s);
  xc[(size_t)idx]   = r;
  xcbf[(size_t)idx] = f2bf(r);
}

// ---- selective scan: one thread per (b, channel), 16-state recurrence ------
__global__ __launch_bounds__(256)
void scan_kernel(const float* __restrict__ xc, const float* __restrict__ dtf,
                 const float* __restrict__ xdbl, const float* __restrict__ Alog,
                 const float* __restrict__ Dvec, const float* __restrict__ xz,
                 unsigned short* __restrict__ ybf, int dir) {
  __shared__ float sBC[2][32];
  int b = blockIdx.x >> 2;
  int c = (blockIdx.x & 3) * 256 + threadIdx.x;    // 0..1023
  float Arow[16], h[16];
#pragma unroll
  for (int s = 0; s < 16; ++s) { Arow[s] = -__expf(Alog[c * 16 + s]); h[s] = 0.f; }
  float Dc = Dvec[c];
  for (int t = 0; t < 1024; ++t) {
    size_t row = (size_t)b * 1024 + t;
    if (threadIdx.x < 32) sBC[t & 1][threadIdx.x] = xdbl[row * 64 + 32 + threadIdx.x];
    __syncthreads();
    const float* Bv = sBC[t & 1];
    const float* Cv = sBC[t & 1] + 16;
    float dtv = dtf[row * 1024 + c];
    float u   = xc [row * 1024 + c];
    float du  = dtv * u;
    float yv  = 0.f;
#pragma unroll
    for (int s = 0; s < 16; ++s) {
      float dA = __expf(dtv * Arow[s]);
      h[s] = dA * h[s] + du * Bv[s];
      yv  += h[s] * Cv[s];
    }
    yv += u * Dc;
    size_t orow = (dir == 0) ? row : ((size_t)b * 1024 + (1023 - t));  // un-flip on write
    float z = xz[orow * 2048 + 1024 + c];
    ybf[orow * 1024 + c] = f2bf(yv * silu_f(z));
  }
}

// ---- out = LayerNorm(a + b) * g + beta ; one wave per 512-wide row ---------
__global__ __launch_bounds__(256)
void add_ln_kernel(const float* __restrict__ a, const float* __restrict__ b,
                   const float* __restrict__ g, const float* __restrict__ beta,
                   float* __restrict__ out, unsigned short* __restrict__ outbf) {
  int wave = threadIdx.x >> 5;
  int lane = threadIdx.x & 31;
  size_t row = (size_t)blockIdx.x * 8 + wave;   // 8192 rows, grid 1024
  const float* pa = a + row * 512;
  const float* pb = b + row * 512;
  float v[16], sum = 0.f, sq = 0.f;
#pragma unroll
  for (int i = 0; i < 16; ++i) {
    int col = lane + i * 32;
    float x = pa[col] + pb[col];
    v[i] = x; sum += x; sq += x * x;
  }
#pragma unroll
  for (int o = 16; o > 0; o >>= 1) {
    sum += __shfl_xor(sum, o, 32);
    sq  += __shfl_xor(sq,  o, 32);
  }
  float mean = sum * (1.f / 512.f);
  float var  = sq * (1.f / 512.f) - mean * mean;
  float inv  = rsqrtf(var + 1e-5f);
#pragma unroll
  for (int i = 0; i < 16; ++i) {
    int col = lane + i * 32;
    float o2 = (v[i] - mean) * inv * g[col] + beta[col];
    out[row * 512 + col] = o2;
    if (outbf) outbf[row * 512 + col] = f2bf(o2);
  }
}

// ---------------------------------------------------------------------------
extern "C" void kernel_launch(void* const* d_in, const int* in_sizes, int n_in,
                              void* d_out, int out_size, void* d_ws, size_t ws_size,
                              hipStream_t stream) {
  const float* x       = (const float*)d_in[0];
  const float* in1_w   = (const float*)d_in[1];
  const float* conv1_w = (const float*)d_in[2];
  const float* conv1_b = (const float*)d_in[3];
  const float* xp1_w   = (const float*)d_in[4];
  const float* dtp1_w  = (const float*)d_in[5];
  const float* dtp1_b  = (const float*)d_in[6];
  const float* Alog1   = (const float*)d_in[7];
  const float* D1      = (const float*)d_in[8];
  const float* out1_w  = (const float*)d_in[9];
  const float* in2_w   = (const float*)d_in[10];
  const float* conv2_w = (const float*)d_in[11];
  const float* conv2_b = (const float*)d_in[12];
  const float* xp2_w   = (const float*)d_in[13];
  const float* dtp2_w  = (const float*)d_in[14];
  const float* dtp2_b  = (const float*)d_in[15];
  const float* Alog2   = (const float*)d_in[16];
  const float* D2      = (const float*)d_in[17];
  const float* out2_w  = (const float*)d_in[18];
  const float* c1_w    = (const float*)d_in[19];
  const float* c1_b    = (const float*)d_in[20];
  const float* c2_w    = (const float*)d_in[21];
  const float* c2_b    = (const float*)d_in[22];
  const float* ln1_g   = (const float*)d_in[23];
  const float* ln1_b   = (const float*)d_in[24];
  const float* ln2_g   = (const float*)d_in[25];
  const float* ln2_b   = (const float*)d_in[26];

  const size_t Mr = 8192;            // B*L
  float* ws   = (float*)d_ws;
  // ---- f32 region ----
  float* xz1  = ws;                         // 8192*2048
  float* xz2  = xz1  + Mr * 2048;
  float* xc1  = xz2  + Mr * 2048;           // 8192*1024
  float* xc2  = xc1  + Mr * 1024;
  float* xd1  = xc2  + Mr * 1024;           // 8192*64
  float* xd2  = xd1  + Mr * 64;
  float* dt1  = xd2  + Mr * 64;             // 8192*1024
  float* dt2  = dt1  + Mr * 1024;
  float* osum = dt2  + Mr * 1024;           // 8192*512
  float* xln  = osum + Mr * 512;
  float* ffo  = xln  + Mr * 512;
  // ---- bf16 region ----
  unsigned short* hb    = (unsigned short*)(ffo + Mr * 512);
  unsigned short* xbf   = hb;               hb += Mr * 512;
  unsigned short* w_in1 = hb;               hb += 2048 * 512;
  unsigned short* w_in2 = hb;               hb += 2048 * 512;
  unsigned short* w_xp1 = hb;               hb += 64 * 1024;
  unsigned short* w_xp2 = hb;               hb += 64 * 1024;
  unsigned short* w_dt1 = hb;               hb += 1024 * 32;
  unsigned short* w_dt2 = hb;               hb += 1024 * 32;
  unsigned short* w_o1  = hb;               hb += 512 * 1024;
  unsigned short* w_o2  = hb;               hb += 512 * 1024;
  unsigned short* w_c1  = hb;               hb += 2048 * 512;
  unsigned short* w_c2  = hb;               hb += 512 * 2048;
  unsigned short* xc1bf = hb;               hb += Mr * 1024;
  unsigned short* xc2bf = hb;               hb += Mr * 1024;
  unsigned short* xd1bf = hb;               hb += Mr * 64;
  unsigned short* xd2bf = hb;               hb += Mr * 64;
  unsigned short* y1bf  = hb;               hb += Mr * 1024;
  unsigned short* y2bf  = hb;               hb += Mr * 1024;
  unsigned short* xlnbf = hb;               hb += Mr * 512;
  unsigned short* hffbf = hb;               hb += Mr * 2048;

  dim3 blk(256);
  auto cvt = [&](const float* src, unsigned short* dst, int n) {
    cvt_f32_bf16<<<dim3(n / 2048), blk, 0, stream>>>(src, dst, n);
  };
  // one-time conversions (weights + input activations)
  cvt(x,      xbf,   8192 * 512);
  cvt(in1_w,  w_in1, 2048 * 512);
  cvt(in2_w,  w_in2, 2048 * 512);
  cvt(xp1_w,  w_xp1, 64 * 1024);
  cvt(xp2_w,  w_xp2, 64 * 1024);
  cvt(dtp1_w, w_dt1, 1024 * 32);
  cvt(dtp2_w, w_dt2, 1024 * 32);
  cvt(out1_w, w_o1,  512 * 1024);
  cvt(out2_w, w_o2,  512 * 1024);
  cvt(c1_w,   w_c1,  2048 * 512);
  cvt(c2_w,   w_c2,  512 * 2048);

  // in_proj (both directions): (8192x512) x (2048x512)^T
  gemm_bf16_wmma<<<dim3(16, 64), blk, 0, stream>>>(xbf, 512, w_in1, 512, nullptr,
      xz1, nullptr, 2048, 8192, 2048, 512, 0, 0);
  gemm_bf16_wmma<<<dim3(16, 64), blk, 0, stream>>>(xbf, 512, w_in2, 512, nullptr,
      xz2, nullptr, 2048, 8192, 2048, 512, 0, 0);
  // depthwise causal conv + SiLU (dir 1 in flipped coordinates)
  conv_silu_kernel<<<dim3(32768), blk, 0, stream>>>(xz1, conv1_w, conv1_b, xc1, xc1bf, 0);
  conv_silu_kernel<<<dim3(32768), blk, 0, stream>>>(xz2, conv2_w, conv2_b, xc2, xc2bf, 1);
  // x_proj: (8192x1024) x (64x1024)^T -> f32 (scan B/C) + bf16 (dt GEMM A)
  gemm_bf16_wmma<<<dim3(1, 64), blk, 0, stream>>>(xc1bf, 1024, w_xp1, 1024, nullptr,
      xd1, xd1bf, 64, 8192, 64, 1024, 0, 0);
  gemm_bf16_wmma<<<dim3(1, 64), blk, 0, stream>>>(xc2bf, 1024, w_xp2, 1024, nullptr,
      xd2, xd2bf, 64, 8192, 64, 1024, 0, 0);
  // dt_proj + softplus: (8192x32) x (1024x32)^T
  gemm_bf16_wmma<<<dim3(8, 64), blk, 0, stream>>>(xd1bf, 64, w_dt1, 32, dtp1_b,
      dt1, nullptr, 1024, 8192, 1024, 32, 1, 0);
  gemm_bf16_wmma<<<dim3(8, 64), blk, 0, stream>>>(xd2bf, 64, w_dt2, 32, dtp2_b,
      dt2, nullptr, 1024, 8192, 1024, 32, 1, 0);
  // selective scan + D skip + SiLU(z) gate; dir 1 writes output un-flipped
  scan_kernel<<<dim3(32), blk, 0, stream>>>(xc1, dt1, xd1, Alog1, D1, xz1, y1bf, 0);
  scan_kernel<<<dim3(32), blk, 0, stream>>>(xc2, dt2, xd2, Alog2, D2, xz2, y2bf, 1);
  // out_proj sum: osum = y1*W1^T + y2*W2^T
  gemm_bf16_wmma<<<dim3(4, 64), blk, 0, stream>>>(y1bf, 1024, w_o1, 1024, nullptr,
      osum, nullptr, 512, 8192, 512, 1024, 0, 0);
  gemm_bf16_wmma<<<dim3(4, 64), blk, 0, stream>>>(y2bf, 1024, w_o2, 1024, nullptr,
      osum, nullptr, 512, 8192, 512, 1024, 0, 1);
  // LN1(osum + x) -> f32 residual + bf16 FFN operand
  add_ln_kernel<<<dim3(1024), blk, 0, stream>>>(osum, x, ln1_g, ln1_b, xln, xlnbf);
  // FFN: gelu(xln*c1^T + b1) -> bf16 only ; then *c2^T + b2 -> f32
  gemm_bf16_wmma<<<dim3(16, 64), blk, 0, stream>>>(xlnbf, 512, w_c1, 512, c1_b,
      nullptr, hffbf, 2048, 8192, 2048, 512, 2, 0);
  gemm_bf16_wmma<<<dim3(4, 64), blk, 0, stream>>>(hffbf, 2048, w_c2, 2048, c2_b,
      ffo, nullptr, 512, 8192, 512, 2048, 0, 0);
  // LN2(xln + ffo) -> output
  add_ln_kernel<<<dim3(1024), blk, 0, stream>>>(xln, ffo, ln2_g, ln2_b,
      (float*)d_out, nullptr);
}
